// LinearAttention_82111184765368
// MI455X (gfx1250) — matrix operand order repaired and checked by reading, not compiled, not run on hardware
//
#include <hip/hip_runtime.h>
#include <hip/hip_bf16.h>
#include <math.h>

// ---------------- WMMA plumbing (CDNA5 / gfx1250, wave32) ----------------
typedef __attribute__((ext_vector_type(16))) __bf16 v16bf;
typedef __attribute__((ext_vector_type(8)))  float  v8f;

__device__ inline v16bf ld_frag2(const __bf16* p0, const __bf16* p1) {
    union { v16bf v; uint4 u[2]; } f;
    f.u[0] = *reinterpret_cast<const uint4*>(p0);
    f.u[1] = *reinterpret_cast<const uint4*>(p1);
    return f.v;
}
// A-matrix fragment (16x32): lane row = l&15, khalf = l>>4 -> K in {kh*8..+8} U {16+kh*8..+8}
__device__ inline v16bf ld_a(const __bf16* row, int kb, int hs) {
    return ld_frag2(row + kb + hs * 8, row + kb + 16 + hs * 8);
}
// B-matrix fragment (32x16): lane col = l&15, khalf = l>>4 -> K in {kh*8..+8} U {16+kh*8..+8}
__device__ inline v16bf ld_b(const __bf16* row, int kb, int hs) {
    return ld_frag2(row + kb + hs * 8, row + kb + 16 + hs * 8);
}
__device__ inline v8f wmma_bf(v16bf a, v16bf b, v8f c) {
    return __builtin_amdgcn_wmma_f32_16x16x32_bf16(false, a, false, b, (short)0, c, false, false);
}

// Problem constants
#define BATCH 16
#define CCH   256
#define NHD   8
#define DHD   64
#define HID   512      // NH*DH
#define NPIX  4096     // 64*64
#define NTOK  65536    // BATCH*NPIX

// ---------------- K0: weights fp32 -> bf16 ----------------
__global__ void cvt_weights(const float* wq, const float* wo, __bf16* wqb, __bf16* wob) {
    int i = blockIdx.x * 256 + threadIdx.x;
    if (i < 1536 * 256) wqb[i] = (__bf16)wq[i];
    if (i < 256 * 512)  wob[i] = (__bf16)wo[i];
}

// ---------------- K1: rmsnorm(x)*gain1 and transpose to token-major bf16 ----------------
// grid (128 pixel-tiles, 16 batches), block 256. Tile: 256 channels x 32 pixels.
__global__ void rmsnorm_in(const float* __restrict__ x, const float* __restrict__ g1,
                           __bf16* __restrict__ xn) {
    __shared__ float tile[256][33];
    __shared__ float red[32][8];
    __shared__ float fac[32];
    int t = threadIdx.x;
    int b = blockIdx.y, p0 = blockIdx.x * 32;
    int p = t & 31, cpart = t >> 5;           // 8 channel groups of 32
    const float* xb = x + (size_t)b * CCH * NPIX + p0 + p;
    float ss = 0.f;
    for (int j = 0; j < 32; ++j) {
        int c = cpart * 32 + j;
        float v = xb[(size_t)c * NPIX];
        tile[c][p] = v;
        ss += v * v;
    }
    red[p][cpart] = ss;
    __syncthreads();
    if (t < 32) {
        float tot = 0.f;
        for (int k = 0; k < 8; ++k) tot += red[t][k];
        fac[t] = 16.0f / fmaxf(sqrtf(tot), 1e-12f);   // sqrt(C)=16
    }
    __syncthreads();
    float g = g1[t];                                   // t == channel on write side
    for (int j = 0; j < 32; ++j) {
        size_t tok = (size_t)b * NPIX + p0 + j;
        xn[tok * CCH + t] = (__bf16)(tile[t][j] * fac[j] * g);
    }
}

// ---------------- K2: qkv GEMM [NTOK x 256] * [256 x 1536], fused q-softmax ----------------
// grid (256 token-tiles of 256, 24 o-tiles of 64), block 256 = 8 waves.
// Wave: 32 tok x 64 o (2 row-tiles x 4 col-tiles); fresh fragment locals per k-step,
// fully unrolled so the scheduler hoists load clauses ahead of the WMMA group.
__global__ void __launch_bounds__(256)
qkv_gemm(const __bf16* __restrict__ xn, const __bf16* __restrict__ wq,
         __bf16* __restrict__ q_s, __bf16* __restrict__ k_raw, __bf16* __restrict__ v_raw) {
    int t = threadIdx.x, w = t >> 5, l = t & 31, lr = l & 15, hs = l >> 4;
    size_t tok0 = (size_t)blockIdx.x * 256 + w * 32;
    int obase = blockIdx.y * 64;
    const __bf16* arow0 = xn + (tok0 + lr) * CCH;
    const __bf16* arow1 = arow0 + (size_t)16 * CCH;
    const __bf16* brow[4];
    for (int j = 0; j < 4; ++j) brow[j] = wq + (size_t)(obase + j * 16 + lr) * CCH;

    v8f acc[2][4] = {};
#pragma unroll
    for (int kb = 0; kb < CCH; kb += 32) {
        v16bf a0 = ld_a(arow0, kb, hs);
        v16bf a1 = ld_a(arow1, kb, hs);
#pragma unroll
        for (int j = 0; j < 4; ++j) {
            v16bf bbj = ld_b(brow[j], kb, hs);
            acc[0][j] = wmma_bf(a0, bbj, acc[0][j]);
            acc[1][j] = wmma_bf(a1, bbj, acc[1][j]);
        }
    }

    if (blockIdx.y < 8) {
        // q region: o-tile == head's full d-range -> softmax over d per token row, *SCALE
        for (int rt = 0; rt < 2; ++rt)
            for (int r = 0; r < 8; ++r) {
                float v0[4], m = -1e30f;
                for (int j = 0; j < 4; ++j) { v0[j] = acc[rt][j][r]; m = fmaxf(m, v0[j]); }
                for (int msk = 1; msk < 16; msk <<= 1) m = fmaxf(m, __shfl_xor(m, msk, 16));
                float e[4], s = 0.f;
                for (int j = 0; j < 4; ++j) { e[j] = __expf(v0[j] - m); s += e[j]; }
                for (int msk = 1; msk < 16; msk <<= 1) s += __shfl_xor(s, msk, 16);
                float inv = 0.125f / s;                   // SCALE = 64^-0.5
                size_t tok = tok0 + rt * 16 + r + hs * 8;
                for (int j = 0; j < 4; ++j)
                    q_s[tok * HID + obase + j * 16 + lr] = (__bf16)(e[j] * inv);
            }
    } else {
        __bf16* dst = (blockIdx.y < 16) ? k_raw : v_raw;
        int od = obase - ((blockIdx.y < 16) ? 512 : 1024);
        for (int rt = 0; rt < 2; ++rt)
            for (int r = 0; r < 8; ++r) {
                size_t tok = tok0 + rt * 16 + r + hs * 8;
                for (int j = 0; j < 4; ++j)
                    dst[tok * HID + od + j * 16 + lr] = (__bf16)acc[rt][j][r];
            }
    }
}

// ---------------- K3: k softmax stats over n (4096 tokens + 4 mem) per (b,h,d) ----------------
// grid (8 heads, 16 batches), block 256: 64 d x 4 token-partitions, online max/sum.
__global__ void kstats(const __bf16* __restrict__ k_raw, const float* __restrict__ mem_kv,
                       float* __restrict__ kmax, float* __restrict__ kinv) {
    __shared__ float rm[64][4], rs[64][4];
    int t = threadIdx.x, d = t & 63, part = t >> 6;
    int h = blockIdx.x, b = blockIdx.y;
    const __bf16* base = k_raw + ((size_t)b * NPIX) * HID + h * 64 + d;
    float m = -1e30f, s = 0.f;
    for (int i = 0; i < 1024; ++i) {
        float v = (float)base[(size_t)(part * 1024 + i) * HID];
        float nm = fmaxf(m, v);
        s = s * __expf(m - nm) + __expf(v - nm);
        m = nm;
    }
    rm[d][part] = m; rs[d][part] = s;
    __syncthreads();
    if (t < 64) {
        float M = rm[t][0], S = rs[t][0];
        for (int p2 = 1; p2 < 4; ++p2) {
            float m2 = rm[t][p2], s2 = rs[t][p2];
            float nm = fmaxf(M, m2);
            S = S * __expf(M - nm) + s2 * __expf(m2 - nm); M = nm;
        }
        for (int mm = 0; mm < 4; ++mm) {                  // mem slots
            float v = mem_kv[(h * 64 + t) * 4 + mm];
            float nm = fmaxf(M, v);
            S = S * __expf(M - nm) + __expf(v - nm); M = nm;
        }
        kmax[(b * 8 + h) * 64 + t] = M;
        kinv[(b * 8 + h) * 64 + t] = 1.0f / S;
    }
}

// ---------------- K5: ctxT[e][d] = sum_n v[n,e]*softk[n,d], K = 4 mem + 4096 tokens ----------------
// grid (8 heads, 16 batches), block 256 = 8 waves; wave: 16 e x 32 d.
// Pipelined: next chunk's global loads issued between LDS publish and WMMA group.
__global__ void __launch_bounds__(256)
context_gemm(const __bf16* __restrict__ k_raw, const __bf16* __restrict__ v_raw,
             const float* __restrict__ mem_kv, const float* __restrict__ kmax,
             const float* __restrict__ kinv, __bf16* __restrict__ ctxT) {
    __shared__ __bf16 kT[64][40];     // [d][n-chunk]
    __shared__ __bf16 vT[64][40];     // [e][n-chunk]
    __shared__ float smax[64], sinv[64];
    int t = threadIdx.x, h = blockIdx.x, b = blockIdx.y;
    if (t < 64) { smax[t] = kmax[(b * 8 + h) * 64 + t]; sinv[t] = kinv[(b * 8 + h) * 64 + t]; }
    int w = t >> 5, l = t & 31, lr = l & 15, hs = l >> 4;
    int etile = (w >> 1) * 16, dbase = (w & 1) * 32;
    size_t rowbase = ((size_t)b * NPIX) * HID + h * 64;
    int xdim[8], nn[8];
    for (int r = 0; r < 8; ++r) { int flat = t + r * 256; xdim[r] = flat & 63; nn[r] = flat >> 6; }

    v8f acc[2] = {};
    // prologue: chunk 0 = memory kv slots, zero-padded to 32
    float kreg[8]; __bf16 vreg[8];
    for (int r = 0; r < 8; ++r) {
        if (nn[r] < 4) {
            kreg[r] = mem_kv[(h * 64 + xdim[r]) * 4 + nn[r]];
            vreg[r] = (__bf16)mem_kv[2048 + (h * 64 + xdim[r]) * 4 + nn[r]];
        } else { kreg[r] = -1e30f; vreg[r] = (__bf16)0.f; }   // exp -> 0 weight
    }

    for (int chunk = 0; chunk <= 128; ++chunk) {
        __syncthreads();                                   // prior WMMA reads complete
        for (int r = 0; r < 8; ++r) {
            kT[xdim[r]][nn[r]] = (__bf16)(__expf(kreg[r] - smax[xdim[r]]) * sinv[xdim[r]]);
            vT[xdim[r]][nn[r]] = vreg[r];
        }
        __syncthreads();                                   // tiles published
        if (chunk < 128) {                                 // prefetch next chunk (overlaps WMMA)
            int n0 = chunk * 32;
            for (int r = 0; r < 8; ++r) {
                size_t gi = rowbase + (size_t)(n0 + nn[r]) * HID + xdim[r];
                kreg[r] = (float)k_raw[gi];
                vreg[r] = v_raw[gi];
            }
        }
        v16bf a = ld_a(&vT[etile + lr][0], 0, hs);          // A[m=e][k=n]
        for (int jd = 0; jd < 2; ++jd)                      // B[k=n][col=d]
            acc[jd] = wmma_bf(a, ld_b(&kT[dbase + jd * 16 + lr][0], 0, hs), acc[jd]);
    }
    __bf16* cdst = ctxT + (size_t)(b * 8 + h) * 64 * 64;
    for (int jd = 0; jd < 2; ++jd)
        for (int r = 0; r < 8; ++r) {
            int e = etile + r + hs * 8, d = dbase + jd * 16 + lr;
            cdst[e * 64 + d] = (__bf16)acc[jd][r];          // ctxT[e][d] = context[d][e]
        }
}

// ---------------- K6a: hidden[n, h*64+e] = sum_d q_s[n, h*64+d] * ctx[d][e] ----------------
// grid (16 token-tiles of 256, 8 heads, 16 batches), block 256; wave: 32 tok x 64 e.
// K = 64: all 12 fragments hoisted, 16 back-to-back WMMAs.
__global__ void __launch_bounds__(256)
attn_out(const __bf16* __restrict__ q_s, const __bf16* __restrict__ ctxT,
         __bf16* __restrict__ hidden) {
    int t = threadIdx.x, w = t >> 5, l = t & 31, lr = l & 15, hs = l >> 4;
    int h = blockIdx.y, b = blockIdx.z;
    size_t tok0 = (size_t)b * NPIX + blockIdx.x * 256 + w * 32;
    const __bf16* ctxh = ctxT + (size_t)(b * 8 + h) * 64 * 64;
    const __bf16* arow0 = q_s + (tok0 + lr) * HID + h * 64;
    const __bf16* arow1 = arow0 + (size_t)16 * HID;

    v16bf a[2][2], bfr[4][2];
#pragma unroll
    for (int kk = 0; kk < 2; ++kk) {
        a[0][kk] = ld_a(arow0, kk * 32, hs);
        a[1][kk] = ld_a(arow1, kk * 32, hs);
        for (int j = 0; j < 4; ++j)
            bfr[j][kk] = ld_b(ctxh + (size_t)(j * 16 + lr) * 64, kk * 32, hs);
    }
    v8f acc[2][4] = {};
#pragma unroll
    for (int kk = 0; kk < 2; ++kk)
        for (int rt = 0; rt < 2; ++rt)
            for (int j = 0; j < 4; ++j)
                acc[rt][j] = wmma_bf(a[rt][kk], bfr[j][kk], acc[rt][j]);

    for (int rt = 0; rt < 2; ++rt)
        for (int r = 0; r < 8; ++r) {
            size_t tok = tok0 + rt * 16 + r + hs * 8;
            for (int j = 0; j < 4; ++j)
                hidden[tok * HID + h * 64 + j * 16 + lr] = (__bf16)acc[rt][j][r];
        }
}

// ---------------- K6b: out = rmsnorm(hidden @ w_out^T + b_out) -> (B,C,H,W) ----------------
// grid (128 token-tiles of 32, 16 batches), block 256; wave: 16 tok x 64 c; fused rmsnorm.
__global__ void __launch_bounds__(256)
out_proj(const __bf16* __restrict__ hidden, const __bf16* __restrict__ wo,
         const float* __restrict__ bo, const float* __restrict__ g2,
         float* __restrict__ out) {
    __shared__ float sfin[32][257];
    __shared__ float red[32][8];
    __shared__ float fac[32];
    int t = threadIdx.x, w = t >> 5, l = t & 31, lr = l & 15, hs = l >> 4;
    int b = blockIdx.y, p0 = blockIdx.x * 32;
    size_t tok0 = (size_t)b * NPIX + p0 + (w & 1) * 16;
    int cbase = (w >> 1) * 64;
    const __bf16* arow = hidden + (tok0 + lr) * HID;
    const __bf16* brow[4];
    for (int j = 0; j < 4; ++j) brow[j] = wo + (size_t)(cbase + j * 16 + lr) * HID;

    v8f acc[4] = {};
#pragma unroll 4
    for (int kb = 0; kb < HID; kb += 32) {
        v16bf a = ld_a(arow, kb, hs);
#pragma unroll
        for (int j = 0; j < 4; ++j)
            acc[j] = wmma_bf(a, ld_b(brow[j], kb, hs), acc[j]);
    }
    for (int r = 0; r < 8; ++r) {
        int tl = (w & 1) * 16 + r + hs * 8;
        for (int j = 0; j < 4; ++j) {
            int c = cbase + j * 16 + lr;
            sfin[tl][c] = acc[j][r] + bo[c];
        }
    }
    __syncthreads();
    {
        int tok = t & 31, part = t >> 5;
        float ss = 0.f;
        for (int m2 = 0; m2 < 32; ++m2) { float v = sfin[tok][part * 32 + m2]; ss += v * v; }
        red[tok][part] = ss;
    }
    __syncthreads();
    if (t < 32) {
        float tot = 0.f;
        for (int p2 = 0; p2 < 8; ++p2) tot += red[t][p2];
        fac[t] = 16.0f / fmaxf(sqrtf(tot), 1e-12f);
    }
    __syncthreads();
    int tok = t & 31, part = t >> 5;
    for (int m2 = 0; m2 < 32; ++m2) {
        int c = part + 8 * m2;
        out[((size_t)b * CCH + c) * NPIX + p0 + tok] = sfin[tok][c] * fac[tok] * g2[c];
    }
}

// ---------------- Launch ----------------
extern "C" void kernel_launch(void* const* d_in, const int* in_sizes, int n_in,
                              void* d_out, int out_size, void* d_ws, size_t ws_size,
                              hipStream_t stream) {
    (void)in_sizes; (void)n_in; (void)out_size; (void)ws_size;
    const float* x      = (const float*)d_in[0];
    const float* gain1  = (const float*)d_in[1];
    const float* w_qkv  = (const float*)d_in[2];
    const float* mem_kv = (const float*)d_in[3];
    const float* w_out  = (const float*)d_in[4];
    const float* b_out  = (const float*)d_in[5];
    const float* gain2  = (const float*)d_in[6];
    float* out = (float*)d_out;

    char* ws = (char*)d_ws;
    __bf16* xn    = (__bf16*)(ws);                 // 65536*256*2  = 32 MiB
    __bf16* wqb   = (__bf16*)(ws +  33554432);     // 1536*256*2
    __bf16* wob   = (__bf16*)(ws +  34340864);     // 256*512*2
    __bf16* q_s   = (__bf16*)(ws +  34603008);     // 65536*512*2 = 64 MiB
    __bf16* k_raw = (__bf16*)(ws + 101711872);     // 64 MiB (reused as `hidden` after K5)
    __bf16* v_raw = (__bf16*)(ws + 168820736);     // 64 MiB
    float*  kmax  = (float*) (ws + 235929600);     // 8192 f32
    float*  kinv  = (float*) (ws + 235962368);     // 8192 f32
    __bf16* ctxT  = (__bf16*)(ws + 235995136);     // 16*8*64*64*2 = 1 MiB
    __bf16* hidden = k_raw;                        // k_raw is dead after context_gemm

    cvt_weights<<<1536, 256, 0, stream>>>(w_qkv, w_out, wqb, wob);
    rmsnorm_in<<<dim3(128, 16), 256, 0, stream>>>(x, gain1, xn);
    qkv_gemm<<<dim3(256, 24), 256, 0, stream>>>(xn, wqb, q_s, k_raw, v_raw);
    kstats<<<dim3(8, 16), 256, 0, stream>>>(k_raw, mem_kv, kmax, kinv);
    context_gemm<<<dim3(8, 16), 256, 0, stream>>>(k_raw, v_raw, mem_kv, kmax, kinv, ctxT);
    attn_out<<<dim3(16, 8, 16), 256, 0, stream>>>(q_s, ctxT, hidden);
    out_proj<<<dim3(128, 16), 256, 0, stream>>>(hidden, wob, b_out, gain2, out);
}
